// LSTMClassifier_11836929868189
// MI455X (gfx1250) — matrix operand order repaired and checked by reading, not compiled, not run on hardware
//
#include <hip/hip_runtime.h>

// ---------------------------------------------------------------------------
// LSTM classifier for MI455X (gfx1250, wave32, WMMA + TDM).
//   B=1024, T=256, F=128, H=512, C=10
//   * Fused bf16 weight Wc = [W_ih | W_hh]  (2048 x 640), bias fused.
//   * Per timestep: gates = [x_t | h_{t-1}] @ Wc^T + b via
//     v_wmma_f32_16x16x32_bf16, fused sigmoid/tanh + c/h update.
//   * Block-uniform B tile (4 gates x 16 cols x 640 K = 80KB) staged into LDS
//     by the Tensor Data Mover (20 chunk descriptors, padded rows: 80B slots
//     -> conflict-free ds_load_b128 fragment reads), waited with TENSORcnt.
//   * A fragments prefetched one k-chunk ahead to overlap global latency.
// ---------------------------------------------------------------------------

#define B_  1024
#define T_  256
#define F_  128
#define H_  512
#define G_  2048          // 4*H
#define K_  640           // F + H
#define XROW (T_ * F_)    // x row stride for fixed t (floats)

#define NCHUNK    20          // 640 / 32
#define CHUNK_B   5120        // bytes per chunk in LDS: 64 rows * 80B
#define CHUNK_E   2560        // elements per chunk (bf16)
#define ROWSLOT_E 40          // padded row slot: 32 data + 8 pad (bf16 elems)

typedef __attribute__((ext_vector_type(16))) __bf16 v16bf;
typedef __attribute__((ext_vector_type(8)))  __bf16 v8bf;
typedef __attribute__((ext_vector_type(8)))  float  v8f;
typedef __attribute__((ext_vector_type(4)))  float  vf4;
typedef __attribute__((ext_vector_type(4)))  unsigned int u32x4;
typedef __attribute__((ext_vector_type(8)))  int    i32x8;
typedef __attribute__((ext_vector_type(4)))  int    i32x4;

static __device__ __forceinline__ v8f wmma_bf16(v16bf a, v16bf b, v8f c) {
  return __builtin_amdgcn_wmma_f32_16x16x32_bf16(false, a, false, b, (short)0, c,
                                                 false, false);
}

// A fragment (16x32 bf16) from row-major bf16 (rows x H_).
static __device__ __forceinline__ v16bf load_a_h(const __bf16* __restrict__ h,
                                                 int row0, int kc, int l15, int half) {
  const __bf16* p = h + (size_t)(row0 + l15) * H_ + kc + half * 8;
  v8bf lo = *(const v8bf*)(p);
  v8bf hi = *(const v8bf*)(p + 16);
  return __builtin_shufflevector(lo, hi, 0,1,2,3,4,5,6,7,8,9,10,11,12,13,14,15);
}

// A fragment from fp32 x_t (row stride T*F), converted to bf16 in-register.
static __device__ __forceinline__ v16bf load_a_x(const float* __restrict__ xt,
                                                 int row0, int kc, int l15, int half) {
  const float* p = xt + (size_t)(row0 + l15) * XROW + kc + half * 8;
  vf4 f0 = *(const vf4*)(p);
  vf4 f1 = *(const vf4*)(p + 4);
  vf4 f2 = *(const vf4*)(p + 16);
  vf4 f3 = *(const vf4*)(p + 20);
  v16bf a;
#pragma unroll
  for (int i = 0; i < 4; ++i) {
    a[i]      = (__bf16)f0[i];
    a[4 + i]  = (__bf16)f1[i];
    a[8 + i]  = (__bf16)f2[i];
    a[12 + i] = (__bf16)f3[i];
  }
  return a;
}

// B fragment (32x16 bf16) from padded LDS chunk image.
// chunk c holds [gate][16 rows][32 K] with 80B row slots.
static __device__ __forceinline__ v16bf load_b_lds(const __bf16* __restrict__ sm,
                                                   int c, int g, int l15, int half) {
  const __bf16* p = sm + (size_t)c * CHUNK_E + (g * 16 + l15) * ROWSLOT_E + half * 8;
  v8bf lo = *(const v8bf*)(p);
  v8bf hi = *(const v8bf*)(p + 16);
  return __builtin_shufflevector(lo, hi, 0,1,2,3,4,5,6,7,8,9,10,11,12,13,14,15);
}

static __device__ __forceinline__ float sigmoidf_(float v) {
  return 1.0f / (1.0f + __expf(-v));
}
static __device__ __forceinline__ float tanhf_(float v) {
  return 2.0f / (1.0f + __expf(-2.0f * v)) - 1.0f;
}

// ---------------------------------------------------------------------------
__global__ __launch_bounds__(256) void prep_kernel(
    const float* __restrict__ W_ih, const float* __restrict__ W_hh,
    const float* __restrict__ b_ih, const float* __restrict__ b_hh,
    __bf16* __restrict__ Wc, float* __restrict__ bcomb,
    __bf16* __restrict__ h0, float* __restrict__ c0) {
  int idx = blockIdx.x * blockDim.x + threadIdx.x;
  int stride = gridDim.x * blockDim.x;
  for (int i = idx; i < G_ * K_; i += stride) {
    int g = i / K_, k = i - g * K_;
    float w = (k < F_) ? W_ih[(size_t)g * F_ + k] : W_hh[(size_t)g * H_ + (k - F_)];
    Wc[i] = (__bf16)w;
  }
  for (int i = idx; i < G_; i += stride) bcomb[i] = b_ih[i] + b_hh[i];
  for (int i = idx; i < B_ * H_; i += stride) {
    h0[i] = (__bf16)0.0f;
    c0[i] = 0.0f;
  }
}

// ---------------------------------------------------------------------------
// One LSTM timestep. Grid: 128 blocks x 256 threads (8 waves).
//   blockIdx>>2 -> j-tile (16 hidden cols, shared by whole block)
//   (blockIdx&3)*8 + wave -> m-tile (32 batch rows per wave)
// ---------------------------------------------------------------------------
__global__ __launch_bounds__(256) void lstm_step_kernel(
    const float* __restrict__ x_t,      // x + t*F, batch row stride T*F
    const __bf16* __restrict__ Wc,      // (2048, 640) bf16
    const float* __restrict__ bcomb,    // (2048)
    const __bf16* __restrict__ h_in,    // (1024, 512) bf16
    __bf16* __restrict__ h_out,         // (1024, 512) bf16
    float* __restrict__ cbuf) {         // (1024, 512) fp32, in place
  __shared__ __align__(128) __bf16 smem[NCHUNK * CHUNK_E];   // 100 KB

  const int lane = threadIdx.x & 31;
  const int l15  = lane & 15;
  const int half = lane >> 4;
  const int wtid = threadIdx.x >> 5;
  const int j0   = (blockIdx.x >> 2) * 16;
  const int m0   = ((blockIdx.x & 3) * 8 + wtid) * 32;

  // ---- stage B tile (4 gates x 16 cols x 640 K) into LDS via TDM ----
#if __has_builtin(__builtin_amdgcn_tensor_load_to_lds)
  if (wtid == 0) {
    unsigned lds_base = (unsigned)(size_t)(&smem[0]);
    unsigned long long gbase =
        (unsigned long long)(const void*)(Wc + (size_t)j0 * K_);
    // group1: wg_mask=0, data_size=2B, pad_enable, pad_interval=16dw,
    //         pad_amount=4dw, tensor_dim0=640, tensor_dim1=2048,
    //         tile = 32 x 16 x 4, stride0=640, stride1=512*640
    i32x8 g1;
    g1[0] = (1 << 16) | (1 << 20) | (3 << 22) | (3 << 25);
    g1[1] = (int)(640u << 16);
    g1[2] = (int)(2048u << 16);
    g1[3] = (int)(32u << 16);
    g1[4] = (int)(16u | (4u << 16));
    g1[5] = (int)640;
    g1[6] = 0;                       // stride0 hi16 | stride1 lo16 (327680)
    g1[7] = (int)(327680u >> 16);    // = 5
    i32x4 g2; g2[0] = 4; g2[1] = 0; g2[2] = 0; g2[3] = 0;   // tensor_dim2
    i32x4 g3; g3[0] = 0; g3[1] = 0; g3[2] = 0; g3[3] = 0;
    i32x8 g4;                                               // unused ext group
#pragma unroll
    for (int i = 0; i < 8; ++i) g4[i] = 0;
#pragma unroll
    for (int c = 0; c < NCHUNK; ++c) {
      unsigned long long ga = gbase + (unsigned)(c * 32) * 2ull;
      u32x4 g0;
      g0[0] = 1u;                                   // count=1
      g0[1] = lds_base + (unsigned)(c * CHUNK_B);   // lds_addr
      g0[2] = (unsigned)ga;
      g0[3] = (unsigned)(ga >> 32) | (2u << 30);    // type=2 ("image")
      __builtin_amdgcn_tensor_load_to_lds(g0, g1, g2, g3, g4, 0);
    }
    __builtin_amdgcn_s_wait_tensorcnt(0);
  }
  __syncthreads();
#else
  // Fallback: cooperative copy into the identical padded layout.
  for (int r = threadIdx.x; r < NCHUNK * 64; r += 256) {
    int c = r >> 6, w = r & 63;          // w = g*16 + n
    int g = w >> 4, n = w & 15;
    const __bf16* src = Wc + (size_t)(g * H_ + j0 + n) * K_ + c * 32;
    __bf16* dst = smem + (size_t)c * CHUNK_E + w * ROWSLOT_E;
#pragma unroll
    for (int i = 0; i < 32; ++i) dst[i] = src[i];
  }
  __syncthreads();
#endif

  v8f acc[2][4] = {};

  // ---- K loop: chunks 0..3 from x (fp32->bf16), 4..19 from h (bf16) ----
  v16bf a0 = load_a_x(x_t, m0, 0, l15, half);
  v16bf a1 = load_a_x(x_t, m0 + 16, 0, l15, half);
#pragma unroll
  for (int c = 0; c < NCHUNK; ++c) {
    v16bf na0, na1;
    if (c + 1 < NCHUNK) {              // prefetch next A chunk
      int kn = (c + 1) * 32;
      if (kn < F_) {
        na0 = load_a_x(x_t, m0, kn, l15, half);
        na1 = load_a_x(x_t, m0 + 16, kn, l15, half);
      } else {
        na0 = load_a_h(h_in, m0, kn - F_, l15, half);
        na1 = load_a_h(h_in, m0 + 16, kn - F_, l15, half);
      }
    }
#pragma unroll
    for (int g = 0; g < 4; ++g) {
      v16bf b = load_b_lds(smem, c, g, l15, half);
      acc[0][g] = wmma_bf16(a0, b, acc[0][g]);
      acc[1][g] = wmma_bf16(a1, b, acc[1][g]);
    }
    if (c + 1 < NCHUNK) { a0 = na0; a1 = na1; }
  }

  // ---- fused bias + activations + cell/hidden update ----
  float bias[4];
#pragma unroll
  for (int g = 0; g < 4; ++g) bias[g] = bcomb[g * H_ + j0 + l15];

  const int col = j0 + l15;
#pragma unroll
  for (int mi = 0; mi < 2; ++mi) {
#pragma unroll
    for (int r = 0; r < 8; ++r) {
      int row = m0 + mi * 16 + half * 8 + r;   // D: lanes16-31 -> M = 8+r
      float iv = sigmoidf_(acc[mi][0][r] + bias[0]);
      float fv = sigmoidf_(acc[mi][1][r] + bias[1]);
      float gv = tanhf_(acc[mi][2][r] + bias[2]);
      float ov = sigmoidf_(acc[mi][3][r] + bias[3]);
      size_t idx = (size_t)row * H_ + col;
      float cn = fv * cbuf[idx] + iv * gv;
      cbuf[idx] = cn;
      h_out[idx] = (__bf16)(ov * tanhf_(cn));
    }
  }
}

// ---------------------------------------------------------------------------
__global__ __launch_bounds__(256) void cls_kernel(
    const __bf16* __restrict__ hT, const float* __restrict__ W_cls,
    const float* __restrict__ b_cls, float* __restrict__ out) {
  int idx = blockIdx.x * blockDim.x + threadIdx.x;
  int b = idx >> 4, c = idx & 15;
  if (b < B_ && c < 10) {
    float s = b_cls[c];
#pragma unroll 8
    for (int k = 0; k < H_; ++k)
      s += (float)hT[(size_t)b * H_ + k] * W_cls[(size_t)c * H_ + k];
    out[b * 10 + c] = s;
  }
}

// ---------------------------------------------------------------------------
extern "C" void kernel_launch(void* const* d_in, const int* in_sizes, int n_in,
                              void* d_out, int out_size, void* d_ws, size_t ws_size,
                              hipStream_t stream) {
  const float* x     = (const float*)d_in[0];
  const float* W_ih  = (const float*)d_in[1];
  const float* b_ih  = (const float*)d_in[2];
  const float* W_hh  = (const float*)d_in[3];
  const float* b_hh  = (const float*)d_in[4];
  const float* W_cls = (const float*)d_in[5];
  const float* b_cls = (const float*)d_in[6];
  float* out = (float*)d_out;

  char* ws = (char*)d_ws;
  __bf16* Wc    = (__bf16*)(ws);               // 2,621,440 B
  float*  bcomb = (float*) (ws + 2621440);     //     8,192 B
  __bf16* hA    = (__bf16*)(ws + 2629632);     // 1,048,576 B
  __bf16* hB    = (__bf16*)(ws + 3678208);     // 1,048,576 B
  float*  cbuf  = (float*) (ws + 4726784);     // 2,097,152 B

  prep_kernel<<<1024, 256, 0, stream>>>(W_ih, W_hh, b_ih, b_hh, Wc, bcomb, hA, cbuf);

  for (int t = 0; t < T_; ++t) {
    const __bf16* hin = (t & 1) ? hB : hA;
    __bf16* hout      = (t & 1) ? hA : hB;
    lstm_step_kernel<<<128, 256, 0, stream>>>(x + (size_t)t * F_, Wc, bcomb,
                                              hin, hout, cbuf);
  }
  cls_kernel<<<64, 256, 0, stream>>>(hA, W_cls, b_cls, out);
}
